// EncoderV2_41120016892640
// MI455X (gfx1250) — matrix operand order repaired and checked by reading, not compiled, not run on hardware
//
#include <hip/hip_runtime.h>
#include <hip/hip_bf16.h>

#define HID 128
#define INDIM 64
#define NREL 10
#define NLAYERS 6
#define LN_EPS 1e-5f

typedef __attribute__((ext_vector_type(16))) __bf16 v16bf;
typedef __attribute__((ext_vector_type(8)))  float  v8f;
typedef __attribute__((ext_vector_type(8)))  unsigned short ushort8;
typedef __attribute__((ext_vector_type(4)))  unsigned int uint32x4;
typedef __attribute__((ext_vector_type(8)))  int int32x8;
typedef __attribute__((ext_vector_type(4)))  int int32x4;

union BFrag { v16bf v; ushort8 h[2]; unsigned short u[16]; };
union FAcc  { v8f v; float f[8]; };

__device__ __forceinline__ unsigned short f32_to_bf16(float f) {
    unsigned int u = __float_as_uint(f);
    u += 0x7fffu + ((u >> 16) & 1u);   // round-to-nearest-even
    return (unsigned short)(u >> 16);
}

// ---- weight prep: f32 [nmats][K][N] -> bf16 transposed [nmats][N][K] ----
__global__ void prep_kernel(const float* __restrict__ src, unsigned short* __restrict__ dst,
                            int K, int Nc, int total) {
    int t = blockIdx.x * blockDim.x + threadIdx.x;
    if (t >= total) return;
    int kn = K * Nc;
    int m = t / kn, rem = t - m * kn;
    int k = rem / Nc, n = rem - k * Nc;
    dst[(size_t)m * kn + (size_t)n * K + k] = f32_to_bf16(src[t]);
}

// ---- per-relation in-degree counts ----
__global__ void count_kernel(const int* __restrict__ ei, const int* __restrict__ et,
                             int* __restrict__ cnt, int ne, int nn) {
    int e = blockIdx.x * blockDim.x + threadIdx.x;
    if (e >= ne) return;
    int dst = ei[ne + e];
    int r   = et[e];
    atomicAdd(&cnt[r * nn + dst], 1);
}

__global__ void inv_kernel(const int* __restrict__ cnt, float* __restrict__ inv, int total) {
    int t = blockIdx.x * blockDim.x + threadIdx.x;
    if (t >= total) return;
    int c = cnt[t];
    inv[t] = 1.0f / (float)(c > 1 ? c : 1);
}

// ---- one wave per edge: scatter h[src]*inv[r][dst] into agg[r][dst] (mean folded in) ----
__global__ __launch_bounds__(256) void scatter_kernel(const float* __restrict__ h,
                                                      const int* __restrict__ ei,
                                                      const int* __restrict__ et,
                                                      const float* __restrict__ inv,
                                                      float* __restrict__ agg,
                                                      int ne, int nn) {
    int wid  = (blockIdx.x * blockDim.x + threadIdx.x) >> 5;
    int lane = threadIdx.x & 31;
    if (wid >= ne) return;
    int src = ei[wid];
    int dst = ei[ne + wid];
    int r   = et[wid];
    float s = inv[(size_t)r * nn + dst];
    const float4 v = *(const float4*)&h[(size_t)src * HID + lane * 4];
    float* a = &agg[((size_t)r * nn + dst) * HID + lane * 4];
    atomicAdd(a + 0, v.x * s);
    atomicAdd(a + 1, v.y * s);
    atomicAdd(a + 2, v.z * s);
    atomicAdd(a + 3, v.w * s);
}

// ---- projection: h = x @ Wp + bp  (K=64); also writes bf16 mirror hb ----
__global__ __launch_bounds__(256) void proj_kernel(const float* __restrict__ x,
                                                   const unsigned short* __restrict__ Wpt,
                                                   const float* __restrict__ bp,
                                                   float* __restrict__ h,
                                                   unsigned short* __restrict__ hb, int nn) {
    __shared__ unsigned short lds_a[64 * INDIM];
    __shared__ float          lds_o[64 * HID];
    int row0 = blockIdx.x * 64;
    int lane = threadIdx.x & 31, w = threadIdx.x >> 5;
    int m0 = (w & 3) * 16, n0 = (w >> 2) * 64;

    FAcc acc[4];
#pragma unroll
    for (int i = 0; i < 4; ++i)
#pragma unroll
        for (int j = 0; j < 8; ++j) acc[i].f[j] = 0.f;

    for (int i = threadIdx.x; i < 64 * INDIM; i += 256) {
        int rr = i >> 6, cc = i & 63;
        int node = row0 + rr;
        float v = (node < nn) ? x[(size_t)node * INDIM + cc] : 0.f;
        lds_a[i] = f32_to_bf16(v);
    }
    __syncthreads();

#pragma unroll
    for (int kt = 0; kt < 2; ++kt) {
        int k0 = kt * 32;
        BFrag a;
        int arow = m0 + (lane & 15);
        int kc1  = k0 + ((lane >> 4) << 3);
        a.h[0] = *(const ushort8*)&lds_a[arow * INDIM + kc1];
        a.h[1] = *(const ushort8*)&lds_a[arow * INDIM + kc1 + 16];
#pragma unroll
        for (int nt = 0; nt < 4; ++nt) {
            BFrag b;
            int n  = n0 + nt * 16 + (lane & 15);
            int kb = k0 + ((lane >> 4) << 3);
            b.h[0] = *(const ushort8*)&Wpt[(size_t)n * INDIM + kb];
            b.h[1] = *(const ushort8*)&Wpt[(size_t)n * INDIM + kb + 16];
            acc[nt].v = __builtin_amdgcn_wmma_f32_16x16x32_bf16(
                false, a.v, false, b.v, (short)0, acc[nt].v, false, false);
        }
    }
    __syncthreads();
#pragma unroll
    for (int nt = 0; nt < 4; ++nt) {
        int col = n0 + nt * 16 + (lane & 15);
        int rb  = m0 + ((lane >> 4) << 3);
#pragma unroll
        for (int j = 0; j < 8; ++j) lds_o[(rb + j) * HID + col] = acc[nt].f[j];
    }
    __syncthreads();
    if (threadIdx.x < 64) {
        int rr = threadIdx.x, node = row0 + rr;
        if (node < nn)
            for (int c = 0; c < HID; ++c) {
                float y = lds_o[rr * HID + c] + bp[c];
                h[(size_t)node * HID + c]  = y;
                hb[(size_t)node * HID + c] = f32_to_bf16(y);
            }
    }
}

// ---- fused RGCN layer ----
// out = hin@Wroot + b + sum_r agg_r@Wr ; LayerNorm; ReLU; +residual.
// Root A-tile staged from bf16 mirror hbin via the Tensor Data Mover.
__global__ __launch_bounds__(256) void layer_kernel(const float* __restrict__ hin,
                                                    const unsigned short* __restrict__ hbin,
                                                    float* __restrict__ hout,
                                                    unsigned short* __restrict__ hbout,
                                                    const unsigned short* __restrict__ Rt,
                                                    const unsigned short* __restrict__ Vt,
                                                    const float* __restrict__ agg,
                                                    const float* __restrict__ cb,
                                                    const float* __restrict__ g,
                                                    const float* __restrict__ bln,
                                                    int nn) {
    __shared__ unsigned short lds_a[64 * HID];
    __shared__ float          lds_o[64 * HID];
    int row0 = blockIdx.x * 64;
    int lane = threadIdx.x & 31, w = threadIdx.x >> 5;
    int m0 = (w & 3) * 16, n0 = (w >> 2) * 64;

    FAcc acc[4];
#pragma unroll
    for (int i = 0; i < 4; ++i)
#pragma unroll
        for (int j = 0; j < 8; ++j) acc[i].f[j] = 0.f;

    for (int mat = 0; mat < 1 + NREL; ++mat) {
        __syncthreads();  // protect lds_a vs previous reads
        if (mat == 0) {
            // --- TDM: DMA 64x128 bf16 tile hbin[row0..row0+63][:] -> lds_a ---
            if (w == 0) {
                unsigned long long ga =
                    (unsigned long long)(uintptr_t)(hbin + (size_t)row0 * HID);
                unsigned int galo = (unsigned int)ga;
                unsigned int gahi = (unsigned int)(ga >> 32);
                uint32x4 g0 = {
                    1u,                                   // count=1 (valid D#)
                    (unsigned int)(uintptr_t)lds_a,       // lds_addr (bytes)
                    galo,                                 // global_addr[31:0]
                    (gahi & 0x01FFFFFFu) | (2u << 30)     // global_addr[56:32] | type=2
                };
                int32x8 g1 = {
                    (1 << 16),                            // data_size=1 (2B bf16), no mask
                    (HID & 0xFFFF) << 16,                 // tensor_dim0[15:0]=128
                    ((nn & 0xFFFF) << 16),                // dim0 hi=0 | tensor_dim1 lo=nn
                    ((nn >> 16) & 0xFFFF) | (HID << 16),  // dim1 hi | tile_dim0=128
                    64,                                   // tile_dim1=64, tile_dim2=0
                    HID,                                  // tensor_dim0_stride=128
                    0, 0
                };
                int32x4 z4 = { 0, 0, 0, 0 };
                int32x8 z8 = { 0, 0, 0, 0, 0, 0, 0, 0 };
                __builtin_amdgcn_tensor_load_to_lds(g0, g1, z4, z4, z8, 0);
                __builtin_amdgcn_s_wait_tensorcnt(0);
            }
        } else {
            int r = mat - 1;
            for (int i = threadIdx.x; i < 64 * HID; i += 256) {
                int rr = i >> 7, cc = i & 127;
                int node = row0 + rr;
                float v = (node < nn)
                              ? agg[((size_t)r * nn + node) * HID + cc]
                              : 0.f;
                lds_a[i] = f32_to_bf16(v);
            }
        }
        __syncthreads();
        const unsigned short* Wt = (mat == 0) ? Rt : (Vt + (size_t)(mat - 1) * HID * HID);
        __builtin_prefetch(Wt, 0, 0);
#pragma unroll
        for (int kt = 0; kt < 4; ++kt) {
            int k0 = kt * 32;
            BFrag a;
            int arow = m0 + (lane & 15);
            int kc1  = k0 + ((lane >> 4) << 3);
            a.h[0] = *(const ushort8*)&lds_a[arow * HID + kc1];
            a.h[1] = *(const ushort8*)&lds_a[arow * HID + kc1 + 16];
#pragma unroll
            for (int nt = 0; nt < 4; ++nt) {
                BFrag b;
                int n  = n0 + nt * 16 + (lane & 15);
                int kb = k0 + ((lane >> 4) << 3);
                b.h[0] = *(const ushort8*)&Wt[(size_t)n * HID + kb];
                b.h[1] = *(const ushort8*)&Wt[(size_t)n * HID + kb + 16];
                acc[nt].v = __builtin_amdgcn_wmma_f32_16x16x32_bf16(
                    false, a.v, false, b.v, (short)0, acc[nt].v, false, false);
            }
        }
    }
    __syncthreads();
#pragma unroll
    for (int nt = 0; nt < 4; ++nt) {
        int col = n0 + nt * 16 + (lane & 15);
        int rb  = m0 + ((lane >> 4) << 3);
#pragma unroll
        for (int j = 0; j < 8; ++j) lds_o[(rb + j) * HID + col] = acc[nt].f[j];
    }
    __syncthreads();
    // LayerNorm + ReLU + residual (one thread per row)
    if (threadIdx.x < 64) {
        int rr = threadIdx.x, node = row0 + rr;
        if (node < nn) {
            float mu = 0.f;
            for (int c = 0; c < HID; ++c) {
                float v = lds_o[rr * HID + c] + cb[c];
                lds_o[rr * HID + c] = v;
                mu += v;
            }
            mu *= (1.0f / HID);
            float var = 0.f;
            for (int c = 0; c < HID; ++c) {
                float d = lds_o[rr * HID + c] - mu;
                var += d * d;
            }
            var *= (1.0f / HID);
            float is = rsqrtf(var + LN_EPS);
            for (int c = 0; c < HID; ++c) {
                float y = (lds_o[rr * HID + c] - mu) * is * g[c] + bln[c];
                y = fmaxf(y, 0.f) + hin[(size_t)node * HID + c];
                hout[(size_t)node * HID + c]  = y;
                hbout[(size_t)node * HID + c] = f32_to_bf16(y);
            }
        }
    }
}

// ---- global add pool ----
__global__ void pool_kernel(const float* __restrict__ h, const int* __restrict__ batch,
                            float* __restrict__ out, int nn) {
    int t = blockIdx.x * blockDim.x + threadIdx.x;
    int node = t >> 5, q = t & 31;
    if (node >= nn) return;
    int gph = batch[node];
    const float4 v = *(const float4*)&h[(size_t)node * HID + q * 4];
    float* o = &out[(size_t)gph * HID + q * 4];
    atomicAdd(o + 0, v.x);
    atomicAdd(o + 1, v.y);
    atomicAdd(o + 2, v.z);
    atomicAdd(o + 3, v.w);
}

extern "C" void kernel_launch(void* const* d_in, const int* in_sizes, int n_in,
                              void* d_out, int out_size, void* d_ws, size_t ws_size,
                              hipStream_t stream) {
    const float* x      = (const float*)d_in[0];
    const int*   ei     = (const int*)d_in[1];
    const int*   et     = (const int*)d_in[2];
    const int*   batch  = (const int*)d_in[3];
    const float* Wp     = (const float*)d_in[4];
    const float* bp     = (const float*)d_in[5];
    const float* rel_w  = (const float*)d_in[6];
    const float* root_w = (const float*)d_in[7];
    const float* conv_b = (const float*)d_in[8];
    const float* ln_g   = (const float*)d_in[9];
    const float* ln_b   = (const float*)d_in[10];

    const int nn = in_sizes[0] / INDIM;   // 50000
    const int ne = in_sizes[2];           // 600000

    char* ws = (char*)d_ws;
    size_t off = 0;
    auto take = [&](size_t bytes) -> void* {
        void* p = ws + off;
        off += (bytes + 255) & ~(size_t)255;
        return p;
    };
    float*          agg  = (float*)take((size_t)NREL * nn * HID * 4);
    float*          inv  = (float*)take((size_t)NREL * nn * 4);
    int*            cnt  = (int*)take((size_t)NREL * nn * 4);
    float*          hA   = (float*)take((size_t)nn * HID * 4);
    float*          hB   = (float*)take((size_t)nn * HID * 4);
    unsigned short* hbA  = (unsigned short*)take((size_t)nn * HID * 2);
    unsigned short* hbB  = (unsigned short*)take((size_t)nn * HID * 2);
    unsigned short* Wpt  = (unsigned short*)take((size_t)INDIM * HID * 2);
    unsigned short* Rt   = (unsigned short*)take((size_t)NLAYERS * HID * HID * 2);
    unsigned short* Vt   = (unsigned short*)take((size_t)NLAYERS * NREL * HID * HID * 2);

    // weight prep (bf16 + transpose)
    {
        int tot = INDIM * HID;
        prep_kernel<<<(tot + 255) / 256, 256, 0, stream>>>(Wp, Wpt, INDIM, HID, tot);
        tot = NLAYERS * HID * HID;
        prep_kernel<<<(tot + 255) / 256, 256, 0, stream>>>(root_w, Rt, HID, HID, tot);
        tot = NLAYERS * NREL * HID * HID;
        prep_kernel<<<(tot + 255) / 256, 256, 0, stream>>>(rel_w, Vt, HID, HID, tot);
    }

    // per-relation in-degree -> inverse counts (topology only, once)
    (void)hipMemsetAsync(cnt, 0, (size_t)NREL * nn * 4, stream);
    count_kernel<<<(ne + 255) / 256, 256, 0, stream>>>(ei, et, cnt, ne, nn);
    {
        int tot = NREL * nn;
        inv_kernel<<<(tot + 255) / 256, 256, 0, stream>>>(cnt, inv, tot);
    }

    // projection
    proj_kernel<<<(nn + 63) / 64, 256, 0, stream>>>(x, Wpt, bp, hA, hbA, nn);

    float*          hin   = hA;
    float*          hout  = hB;
    unsigned short* hbin  = hbA;
    unsigned short* hbout = hbB;
    for (int l = 0; l < NLAYERS; ++l) {
        (void)hipMemsetAsync(agg, 0, (size_t)NREL * nn * HID * 4, stream);
        int nwaveblocks = (ne * 32 + 255) / 256;
        scatter_kernel<<<nwaveblocks, 256, 0, stream>>>(hin, ei, et, inv, agg, ne, nn);
        layer_kernel<<<(nn + 63) / 64, 256, 0, stream>>>(
            hin, hbin, hout, hbout,
            Rt + (size_t)l * HID * HID,
            Vt + (size_t)l * NREL * HID * HID,
            agg,
            conv_b + (size_t)l * HID, ln_g + (size_t)l * HID, ln_b + (size_t)l * HID,
            nn);
        float* tf = hin; hin = hout; hout = tf;
        unsigned short* tb = hbin; hbin = hbout; hbout = tb;
    }

    (void)hipMemsetAsync(d_out, 0, (size_t)out_size * 4, stream);
    {
        long long tthreads = (long long)nn * 32;
        pool_kernel<<<(int)((tthreads + 255) / 256), 256, 0, stream>>>(hin, batch, (float*)d_out, nn);
    }
}
